// LocationSensitiveAttention_54640573940207
// MI455X (gfx1250) — compile-verified, matrix operand order checked by reading, not run on hardware
//
#include <hip/hip_runtime.h>
#include <hip/hip_bf16.h>
#include <math.h>

#define RNN_DIM   1024
#define EMB_DIM   512
#define ATTN_DIM  128
#define LOC_F     32
#define LOC_K     31
#define B_SZ      64
#define S_SZ      2048
#define TS        16   // S-tile rows per block

typedef __bf16 v16bf __attribute__((ext_vector_type(16)));
typedef __bf16 v8bf  __attribute__((ext_vector_type(8)));
typedef float  v8f   __attribute__((ext_vector_type(8)));

__device__ __forceinline__ unsigned short f32_to_bf16(float f) {
    unsigned int u = __builtin_bit_cast(unsigned int, f);
    u += 0x7FFFu + ((u >> 16) & 1u);   // round-to-nearest-even
    return (unsigned short)(u >> 16);
}

// ---------------------------------------------------------------------------
// Kernel 1: qproj[b][a] = sum_r query[b,0,r]*Wq[r,a] + attn_bias[a]
// ---------------------------------------------------------------------------
__global__ __launch_bounds__(ATTN_DIM)
void qproj_kernel(const float* __restrict__ query, const float* __restrict__ Wq,
                  const float* __restrict__ attn_bias, float* __restrict__ qproj) {
    int b = blockIdx.x;
    int a = threadIdx.x;
    const float* qrow = query + b * RNN_DIM;
    float acc = attn_bias[a];
    for (int r = 0; r < RNN_DIM; ++r)
        acc += qrow[r] * Wq[r * ATTN_DIM + a];
    qproj[b * ATTN_DIM + a] = acc;
}

// ---------------------------------------------------------------------------
// Kernel 2a: shuffle Wv [512,128] f32 -> bf16 B-fragments
// layout: frag[((t*16 + ks)*32 + lane)*16 + j]
//   N = t*16 + (lane&15); base = ks*32 + ((lane>>4)&1)*8
//   K(j) = base + (j&7) + (j<8 ? 0 : 16)
// ---------------------------------------------------------------------------
__global__ __launch_bounds__(256)
void wvfrag_kernel(const float* __restrict__ Wv, unsigned short* __restrict__ frag) {
    int idx = blockIdx.x * 256 + threadIdx.x;          // 0 .. 8*16*32-1
    if (idx >= 8 * 16 * 32) return;
    int lane = idx & 31;
    int ks   = (idx >> 5) & 15;
    int t    = idx >> 9;
    int n    = t * 16 + (lane & 15);
    int base = ks * 32 + ((lane >> 4) & 1) * 8;
    unsigned short* o = frag + idx * 16;
    for (int j = 0; j < 16; ++j) {
        int k = base + (j & 7) + ((j < 8) ? 0 : 16);
        o[j] = f32_to_bf16(Wv[k * ATTN_DIM + n]);
    }
}

// Kernel 2b: same shuffle for Wloc [32,128] (single k-step)
__global__ __launch_bounds__(256)
void wlocfrag_kernel(const float* __restrict__ Wloc, unsigned short* __restrict__ frag) {
    int idx = threadIdx.x;                              // 0 .. 8*32-1
    int lane = idx & 31;
    int t    = idx >> 5;
    int n    = t * 16 + (lane & 15);
    int base = ((lane >> 4) & 1) * 8;
    unsigned short* o = frag + idx * 16;
    for (int j = 0; j < 16; ++j) {
        int k = base + (j & 7) + ((j < 8) ? 0 : 16);
        o[j] = f32_to_bf16(Wloc[k * ATTN_DIM + n]);
    }
}

// ---------------------------------------------------------------------------
// Kernel 3: per (b, 16-row S tile): conv loc feats + WMMA projection + energy
//   energy_out = d_out align region (raw, pre-softmax)
// ---------------------------------------------------------------------------
__global__ __launch_bounds__(256)
void energy_kernel(const float* __restrict__ value,
                   const float* __restrict__ hist,       // last_alignment_energy [B,S]
                   const float* __restrict__ conv_w,     // [32,1,31]
                   const unsigned short* __restrict__ wvfrag,
                   const unsigned short* __restrict__ wlocfrag,
                   const float* __restrict__ qproj,      // [B,128] bias folded in
                   const float* __restrict__ w_score,    // [128]
                   const float* __restrict__ b_score,
                   float* __restrict__ energy_out) {
    __shared__ __align__(16) unsigned short valbf[TS][EMB_DIM + 8]; // bf16 A rows
    __shared__ __align__(16) unsigned short locbf[TS][LOC_F];       // bf16 loc rows
    __shared__ float part[8][32][8];                                // wave partials

    const int b  = blockIdx.y;
    const int s0 = blockIdx.x * TS;
    const int tid = threadIdx.x;

    // --- stage value[b, s0:s0+16, :] as bf16 into LDS (coalesced) ---
    const float* vbase = value + (size_t)(b * S_SZ + s0) * EMB_DIM;
    for (int i = tid; i < TS * EMB_DIM; i += 256) {
        int row = i >> 9;          // /512
        int col = i & (EMB_DIM - 1);
        valbf[row][col] = f32_to_bf16(vbase[row * EMB_DIM + col]);
    }

    // --- location conv features: locfeat[s_local][f], SAME padding ---
    for (int i = tid; i < TS * LOC_F; i += 256) {
        int sl = i >> 5;
        int f  = i & (LOC_F - 1);
        float acc = 0.f;
        int sg0 = s0 + sl - (LOC_K / 2);
        const float* cw = conv_w + f * LOC_K;
        for (int k = 0; k < LOC_K; ++k) {
            int sg = sg0 + k;
            if (sg >= 0 && sg < S_SZ) acc += hist[b * S_SZ + sg] * cw[k];
        }
        locbf[sl][f] = f32_to_bf16(acc);
    }
    __syncthreads();

    // --- WMMA: each wave owns one 16-col tile of ATTN_DIM ---
    const int wv   = tid >> 5;
    const int lane = tid & 31;
    const int mrow = lane & 15;
    const int kb   = ((lane >> 4) & 1) * 8;

    v8f acc = {};

    // loc @ Wloc  (one 16x16x32 step)
    {
        v16bf a;
        v8bf lo = *(const v8bf*)&locbf[mrow][kb];
        v8bf hi = *(const v8bf*)&locbf[mrow][kb + 16];
        for (int j = 0; j < 8; ++j) { a[j] = lo[j]; a[j + 8] = hi[j]; }
        v16bf bf = *(const v16bf*)(wlocfrag + (size_t)(wv * 32 + lane) * 16);
        acc = __builtin_amdgcn_wmma_f32_16x16x32_bf16(false, a, false, bf,
                                                      (short)0, acc, false, false);
    }
    // value @ Wv  (K = 512 -> 16 k-steps)
    for (int ks = 0; ks < 16; ++ks) {
        int base = ks * 32 + kb;
        v16bf a;
        v8bf lo = *(const v8bf*)&valbf[mrow][base];
        v8bf hi = *(const v8bf*)&valbf[mrow][base + 16];
        for (int j = 0; j < 8; ++j) { a[j] = lo[j]; a[j + 8] = hi[j]; }
        v16bf bf = *(const v16bf*)(wvfrag + (size_t)((wv * 16 + ks) * 32 + lane) * 16);
        acc = __builtin_amdgcn_wmma_f32_16x16x32_bf16(false, a, false, bf,
                                                      (short)0, acc, false, false);
    }

    // --- fuse tanh(. + qproj) * w_score per element ---
    const int acol = wv * 16 + (lane & 15);         // N index in ATTN_DIM
    const float qv = qproj[b * ATTN_DIM + acol];
    const float wsc = w_score[acol];
    for (int r = 0; r < 8; ++r)
        part[wv][lane][r] = tanhf(acc[r] + qv) * wsc;
    __syncthreads();

    // --- reduce over ATTN_DIM: row m gets lanes [0,16) r=m (m<8) or lanes [16,32) r=m-8 ---
    if (tid < TS) {
        int m  = tid;
        int rr = m & 7;
        int l0 = (m < 8) ? 0 : 16;
        float sum = b_score[0];
        for (int w = 0; w < 8; ++w)
            for (int l = l0; l < l0 + 16; ++l)
                sum += part[w][l][rr];
        energy_out[b * S_SZ + s0 + m] = sum;
    }
}

// ---------------------------------------------------------------------------
// Kernel 4: in-place softmax over S for each batch row
// ---------------------------------------------------------------------------
__global__ __launch_bounds__(256)
void softmax_kernel(float* __restrict__ align) {
    __shared__ float red[256];
    int b = blockIdx.x;
    int tid = threadIdx.x;
    float* row = align + b * S_SZ;

    float mx = -INFINITY;
    for (int s = tid; s < S_SZ; s += 256) mx = fmaxf(mx, row[s]);
    red[tid] = mx; __syncthreads();
    for (int off = 128; off > 0; off >>= 1) {
        if (tid < off) red[tid] = fmaxf(red[tid], red[tid + off]);
        __syncthreads();
    }
    mx = red[0]; __syncthreads();

    float sum = 0.f;
    for (int s = tid; s < S_SZ; s += 256) {
        float e = __expf(row[s] - mx);
        row[s] = e;
        sum += e;
    }
    red[tid] = sum; __syncthreads();
    for (int off = 128; off > 0; off >>= 1) {
        if (tid < off) red[tid] += red[tid + off];
        __syncthreads();
    }
    float inv = 1.f / red[0]; __syncthreads();

    for (int s = tid; s < S_SZ; s += 256) row[s] *= inv;
}

// ---------------------------------------------------------------------------
// Kernel 5: context[b,e] = sum_s align[b,s] * value[b,s,e]
// ---------------------------------------------------------------------------
__global__ __launch_bounds__(256)
void context_kernel(const float* __restrict__ value, const float* __restrict__ align,
                    float* __restrict__ context) {
    __shared__ float al[S_SZ];
    int b = blockIdx.x;
    int tid = threadIdx.x;
    for (int s = tid; s < S_SZ; s += 256) al[s] = align[b * S_SZ + s];
    __syncthreads();

    float a0 = 0.f, a1 = 0.f;
    int e0 = tid, e1 = tid + 256;
    const float* vb = value + (size_t)b * S_SZ * EMB_DIM;
    for (int s = 0; s < S_SZ; ++s) {
        float w = al[s];
        const float* vr = vb + (size_t)s * EMB_DIM;
        a0 += w * vr[e0];
        a1 += w * vr[e1];
    }
    context[b * EMB_DIM + e0] = a0;
    context[b * EMB_DIM + e1] = a1;
}

// ---------------------------------------------------------------------------
extern "C" void kernel_launch(void* const* d_in, const int* in_sizes, int n_in,
                              void* d_out, int out_size, void* d_ws, size_t ws_size,
                              hipStream_t stream) {
    const float* query     = (const float*)d_in[0];
    const float* value     = (const float*)d_in[1];
    const float* hist      = (const float*)d_in[2];
    const float* Wq        = (const float*)d_in[3];
    const float* Wv        = (const float*)d_in[4];
    const float* conv_w    = (const float*)d_in[5];
    const float* Wloc      = (const float*)d_in[6];
    const float* attn_bias = (const float*)d_in[7];
    const float* w_score   = (const float*)d_in[8];
    const float* b_score   = (const float*)d_in[9];

    float* out     = (float*)d_out;
    float* context = out;                       // [B, EMB_DIM]
    float* align   = out + B_SZ * EMB_DIM;      // [B, S] (raw energy -> softmax in place)

    // workspace layout
    char* ws = (char*)d_ws;
    float* qproj            = (float*)ws;                                  // 32 KB
    unsigned short* wvfrag  = (unsigned short*)(ws + 32768);               // 128 KB
    unsigned short* wlfrag  = (unsigned short*)(ws + 32768 + 131072);      // 8 KB

    qproj_kernel<<<B_SZ, ATTN_DIM, 0, stream>>>(query, Wq, attn_bias, qproj);
    wvfrag_kernel<<<(8 * 16 * 32 + 255) / 256, 256, 0, stream>>>(Wv, wvfrag);
    wlocfrag_kernel<<<1, 256, 0, stream>>>(Wloc, wlfrag);

    dim3 grid(S_SZ / TS, B_SZ);
    energy_kernel<<<grid, 256, 0, stream>>>(value, hist, conv_w, wvfrag, wlfrag,
                                            qproj, w_score, b_score, align);
    softmax_kernel<<<B_SZ, 256, 0, stream>>>(align);
    context_kernel<<<B_SZ, 256, 0, stream>>>(value, align, context);
}
